// GraphTransformerLayer_40295383171717
// MI455X (gfx1250) — compile-verified, matrix operand order checked by reading, not run on hardware
//
#include <hip/hip_runtime.h>

typedef __bf16 v16bf __attribute__((ext_vector_type(16)));
typedef float  v8f   __attribute__((ext_vector_type(8)));

__device__ __forceinline__ unsigned short f2bf(float f) {
  unsigned u = __float_as_uint(f);
  u += 0x7fffu + ((u >> 16) & 1u);          // round-to-nearest-even
  return (unsigned short)(u >> 16);
}
__device__ __forceinline__ __bf16 bits2bf(unsigned u) {
  return __builtin_bit_cast(__bf16, (unsigned short)(u & 0xffffu));
}
__device__ __forceinline__ void atomicMaxF(float* addr, float val) {
  if (val >= 0.0f) atomicMax((int*)addr, __float_as_int(val));
  else             atomicMin((unsigned int*)addr, __float_as_uint(val));
}

// ---------------- weight convert: W[k][ncol] f32 -> Wt[n][k] bf16 ----------------
__global__ void wt_convert(const float* __restrict__ W, unsigned short* __restrict__ Wt,
                           int K, int Ncol) {
  long long gid = (long long)blockIdx.x * blockDim.x + threadIdx.x;
  long long total = (long long)K * Ncol;
  if (gid >= total) return;
  int n = (int)(gid / K);
  int k = (int)(gid % K);
  Wt[gid] = f2bf(W[(long long)k * Ncol + n]);
}

__global__ void fillf4(float4* __restrict__ p, float v, long long n4) {
  long long gid = (long long)blockIdx.x * blockDim.x + threadIdx.x;
  if (gid < n4) { float4 x = {v, v, v, v}; p[gid] = x; }
}

// ---------------- generic GEMM: C[M,128] = A[M,128] @ W[128,128] (+bias)(+resid) ----------------
template <bool HAS_BIAS, bool HAS_RESID>
__global__ __launch_bounds__(256)
void gemm128_bf16(const float* __restrict__ A, const unsigned* __restrict__ Wt,
                  const float* __restrict__ bias, const float* __restrict__ resid,
                  float* __restrict__ C, int M) {
  __shared__ __align__(16) unsigned shA[8192];   // 128x128 bf16
  __shared__ __align__(16) unsigned shW[8192];   // [n][k] bf16
  int tid = threadIdx.x;
  int mb = blockIdx.x * 128;
  bool fullTile = (mb + 128 <= M);

  { // W tile: 2048 uint4
    const uint4* W4 = (const uint4*)Wt;
    uint4* S4 = (uint4*)shW;
    for (int i = tid; i < 2048; i += 256) S4[i] = W4[i];
  }
  { // A tile: float4 global loads -> packed bf16 pairs
    const float4* A4 = (const float4*)A;
    for (int i = tid; i < 4096; i += 256) {
      int r = i >> 5;              // row in tile
      int kq = (i & 31) << 2;      // k base (multiple of 4)
      int row = mb + r;
      uint2 p = {0u, 0u};
      if (row < M) {
        float4 f = A4[((long long)row * 128 + kq) >> 2];
        p.x = (unsigned)f2bf(f.x) | ((unsigned)f2bf(f.y) << 16);
        p.y = (unsigned)f2bf(f.z) | ((unsigned)f2bf(f.w) << 16);
      }
      *(uint2*)&shA[(r * 128 + kq) >> 1] = p;
    }
  }
  __syncthreads();

  int lane = tid & 31, w = tid >> 5;
  int rb = w * 16;
  int halfsel = lane >> 4;            // K+8 half select (ISA 16-bit A layout)
  int lrow = rb + (lane & 15);
  int ln = lane & 15;

  v16bf afrag[4];
#pragma unroll
  for (int kk = 0; kk < 4; ++kk)
#pragma unroll
    for (int v = 0; v < 8; ++v) {
      int kb = kk * 32 + ((v >> 2) << 4) + (halfsel << 3) + ((v & 3) << 1);
      unsigned u = shA[(lrow * 128 + kb) >> 1];
      afrag[kk][2 * v] = bits2bf(u);
      afrag[kk][2 * v + 1] = bits2bf(u >> 16);
    }

#pragma unroll
  for (int nt = 0; nt < 8; ++nt) {
    v8f acc = {0.f, 0.f, 0.f, 0.f, 0.f, 0.f, 0.f, 0.f};
    int nrow = nt * 16 + ln;
#pragma unroll
    for (int kk = 0; kk < 4; ++kk) {
      v16bf bfrag;
#pragma unroll
      for (int v = 0; v < 8; ++v) {
        int kb = kk * 32 + ((v >> 2) << 4) + (halfsel << 3) + ((v & 3) << 1);
        unsigned u = shW[(nrow * 128 + kb) >> 1];
        bfrag[2 * v] = bits2bf(u);
        bfrag[2 * v + 1] = bits2bf(u >> 16);
      }
      acc = __builtin_amdgcn_wmma_f32_16x16x32_bf16(false, afrag[kk], false, bfrag,
                                                    (short)0, acc, false, false);
    }
    int n = nt * 16 + ln;
    float bval = HAS_BIAS ? bias[n] : 0.0f;
    if (fullTile) {
#pragma unroll
      for (int v = 0; v < 8; ++v) {
        long long idx = (long long)(mb + rb + v + halfsel * 8) * 128 + n;
        float val = acc[v] + bval;
        if (HAS_RESID) val += resid[idx];
        C[idx] = val;
      }
    } else {
#pragma unroll
      for (int v = 0; v < 8; ++v) {
        int grow = mb + rb + v + halfsel * 8;
        if (grow < M) {
          long long idx = (long long)grow * 128 + n;
          float val = acc[v] + bval;
          if (HAS_RESID) val += resid[idx];
          C[idx] = val;
        }
      }
    }
  }
}

// ---------------- fused FFN: out = relu(A@W1+b1)@W2 + b2 + A, mid kept in LDS ----------------
__global__ __launch_bounds__(256)
void ffn_fused(const float* __restrict__ A, const unsigned* __restrict__ W1t,
               const float* __restrict__ b1, const unsigned* __restrict__ W2t,
               const float* __restrict__ b2, float* __restrict__ out, int M) {
  __shared__ __align__(16) unsigned shA[8192];     // 32 KB: A tile 128x128 bf16
  __shared__ __align__(16) unsigned shW[16384];    // 64 KB: W1t then W2t
  __shared__ __align__(16) unsigned shMid[16384];  // 64 KB: mid 128x256 bf16
  int tid = threadIdx.x;
  int mb = blockIdx.x * 128;
  bool fullTile = (mb + 128 <= M);

  {
    const uint4* W4 = (const uint4*)W1t;
    uint4* S4 = (uint4*)shW;
    for (int i = tid; i < 4096; i += 256) S4[i] = W4[i];
  }
  {
    const float4* A4 = (const float4*)A;
    for (int i = tid; i < 4096; i += 256) {
      int r = i >> 5;
      int kq = (i & 31) << 2;
      int row = mb + r;
      uint2 p = {0u, 0u};
      if (row < M) {
        float4 f = A4[((long long)row * 128 + kq) >> 2];
        p.x = (unsigned)f2bf(f.x) | ((unsigned)f2bf(f.y) << 16);
        p.y = (unsigned)f2bf(f.z) | ((unsigned)f2bf(f.w) << 16);
      }
      *(uint2*)&shA[(r * 128 + kq) >> 1] = p;
    }
  }
  __syncthreads();

  int lane = tid & 31, w = tid >> 5;
  int rb = w * 16;
  int halfsel = lane >> 4;
  int lrow = rb + (lane & 15);
  int ln = lane & 15;

  v16bf afrag[4];
#pragma unroll
  for (int kk = 0; kk < 4; ++kk)
#pragma unroll
    for (int v = 0; v < 8; ++v) {
      int kb = kk * 32 + ((v >> 2) << 4) + (halfsel << 3) + ((v & 3) << 1);
      unsigned u = shA[(lrow * 128 + kb) >> 1];
      afrag[kk][2 * v] = bits2bf(u);
      afrag[kk][2 * v + 1] = bits2bf(u >> 16);
    }

  unsigned short* mid16 = (unsigned short*)shMid;
#pragma unroll
  for (int nt = 0; nt < 16; ++nt) {          // phase 1: 128 -> 256, relu
    v8f acc = {0.f, 0.f, 0.f, 0.f, 0.f, 0.f, 0.f, 0.f};
    int nrow = nt * 16 + ln;
#pragma unroll
    for (int kk = 0; kk < 4; ++kk) {
      v16bf bfrag;
#pragma unroll
      for (int v = 0; v < 8; ++v) {
        int kb = kk * 32 + ((v >> 2) << 4) + (halfsel << 3) + ((v & 3) << 1);
        unsigned u = shW[(nrow * 128 + kb) >> 1];
        bfrag[2 * v] = bits2bf(u);
        bfrag[2 * v + 1] = bits2bf(u >> 16);
      }
      acc = __builtin_amdgcn_wmma_f32_16x16x32_bf16(false, afrag[kk], false, bfrag,
                                                    (short)0, acc, false, false);
    }
    int n = nt * 16 + ln;
    float bb = b1[n];
#pragma unroll
    for (int v = 0; v < 8; ++v) {
      int m = rb + v + halfsel * 8;
      float val = fmaxf(acc[v] + bb, 0.0f);
      mid16[m * 256 + n] = f2bf(val);
    }
  }
  __syncthreads();
  {
    const uint4* W4 = (const uint4*)W2t;
    uint4* S4 = (uint4*)shW;
    for (int i = tid; i < 4096; i += 256) S4[i] = W4[i];
  }
  __syncthreads();

  v16bf a2[8];                                // phase 2: 256 -> 128
#pragma unroll
  for (int kk = 0; kk < 8; ++kk)
#pragma unroll
    for (int v = 0; v < 8; ++v) {
      int kb = kk * 32 + ((v >> 2) << 4) + (halfsel << 3) + ((v & 3) << 1);
      unsigned u = shMid[(lrow * 256 + kb) >> 1];
      a2[kk][2 * v] = bits2bf(u);
      a2[kk][2 * v + 1] = bits2bf(u >> 16);
    }
#pragma unroll
  for (int nt = 0; nt < 8; ++nt) {
    v8f acc = {0.f, 0.f, 0.f, 0.f, 0.f, 0.f, 0.f, 0.f};
    int nrow = nt * 16 + ln;
#pragma unroll
    for (int kk = 0; kk < 8; ++kk) {
      v16bf bfrag;
#pragma unroll
      for (int v = 0; v < 8; ++v) {
        int kb = kk * 32 + ((v >> 2) << 4) + (halfsel << 3) + ((v & 3) << 1);
        unsigned u = shW[(nrow * 256 + kb) >> 1];
        bfrag[2 * v] = bits2bf(u);
        bfrag[2 * v + 1] = bits2bf(u >> 16);
      }
      acc = __builtin_amdgcn_wmma_f32_16x16x32_bf16(false, a2[kk], false, bfrag,
                                                    (short)0, acc, false, false);
    }
    int n = nt * 16 + ln;
    float bb = b2[n];
    if (fullTile) {
#pragma unroll
      for (int v = 0; v < 8; ++v) {
        long long idx = (long long)(mb + rb + v + halfsel * 8) * 128 + n;
        out[idx] = acc[v] + bb + A[idx];
      }
    } else {
#pragma unroll
      for (int v = 0; v < 8; ++v) {
        int grow = mb + rb + v + halfsel * 8;
        if (grow < M) {
          long long idx = (long long)grow * 128 + n;
          out[idx] = acc[v] + bb + A[idx];
        }
      }
    }
  }
}

// ---------------- attention: score + segment-max ----------------
__global__ void attn_score(const float* __restrict__ kh, const float* __restrict__ qh,
                           float* __restrict__ eatt, const int* __restrict__ src,
                           const int* __restrict__ dst, float* __restrict__ mbuf,
                           long long total) {
  long long gid = (long long)blockIdx.x * blockDim.x + threadIdx.x;
  if (gid >= total) return;
  int c = (int)(gid & 127);
  int eidx = (int)(gid >> 7);
  int s = src[eidx], d = dst[eidx];
  float dot = kh[(long long)s * 128 + c] * qh[(long long)d * 128 + c];
  dot += __shfl_xor(dot, 8, 16);      // 16-lane head reduction (wave32)
  dot += __shfl_xor(dot, 4, 16);
  dot += __shfl_xor(dot, 2, 16);
  dot += __shfl_xor(dot, 1, 16);
  float sc = dot * 0.25f + __builtin_nontemporal_load(&eatt[gid]); // pe in eatt
  __builtin_nontemporal_store(sc, &eatt[gid]);
  atomicMaxF(&mbuf[(long long)d * 128 + c], sc);
}

__global__ void attn_expsum(const float* __restrict__ eatt, const int* __restrict__ dst,
                            const float* __restrict__ mbuf, float* __restrict__ ssum,
                            long long total) {
  long long gid = (long long)blockIdx.x * blockDim.x + threadIdx.x;
  if (gid >= total) return;
  int c = (int)(gid & 127);
  int d = dst[(int)(gid >> 7)];
  float ex = __expf(__builtin_nontemporal_load(&eatt[gid]) - mbuf[(long long)d * 128 + c]);
  atomicAdd(&ssum[(long long)d * 128 + c], ex);
}

__global__ void attn_agg(const float* __restrict__ eatt, const float* __restrict__ vh,
                         const int* __restrict__ src, const int* __restrict__ dst,
                         const float* __restrict__ mbuf, const float* __restrict__ ssum,
                         float* __restrict__ wv, long long total) {
  long long gid = (long long)blockIdx.x * blockDim.x + threadIdx.x;
  if (gid >= total) return;
  int c = (int)(gid & 127);
  int eidx = (int)(gid >> 7);
  int s = src[eidx], d = dst[eidx];
  float ex = __expf(__builtin_nontemporal_load(&eatt[gid]) - mbuf[(long long)d * 128 + c]);
  float alpha = ex / ssum[(long long)d * 128 + c];
  atomicAdd(&wv[(long long)d * 128 + c], alpha * vh[(long long)s * 128 + c]);
}

// ---------------- batch norm ----------------
__global__ __launch_bounds__(256)
void bn_stats(const float* __restrict__ X, float* __restrict__ gsum,
              float* __restrict__ gsq, int M) {
  const int RPB = 2048;
  int c = threadIdx.x & 127;
  int half = threadIdx.x >> 7;
  long long r0 = (long long)blockIdx.x * RPB + half;
  float s = 0.f, q = 0.f;
  for (int i = 0; i < RPB; i += 2) {
    long long r = r0 + i;
    if (r < M) { float x = X[r * 128 + c]; s += x; q += x * x; }
  }
  __shared__ float sh[512];
  sh[threadIdx.x] = s; sh[256 + threadIdx.x] = q;
  __syncthreads();
  if (threadIdx.x < 128) {
    atomicAdd(&gsum[c], sh[threadIdx.x] + sh[threadIdx.x + 128]);
    atomicAdd(&gsq[c], sh[256 + threadIdx.x] + sh[256 + threadIdx.x + 128]);
  }
}

__global__ void bn_finalize(const float* __restrict__ gsum, const float* __restrict__ gsq,
                            const float* __restrict__ g, const float* __restrict__ b,
                            float* __restrict__ scale, float* __restrict__ shift, float invM) {
  int c = threadIdx.x;
  float mu = gsum[c] * invM;
  float var = gsq[c] * invM - mu * mu;
  float sc = g[c] * rsqrtf(var + 1e-5f);
  scale[c] = sc;
  shift[c] = b[c] - mu * sc;
}

__global__ void bn_apply4(const float4* __restrict__ X, const float* __restrict__ scale,
                          const float* __restrict__ shift, float4* __restrict__ Y,
                          long long n4) {
  long long gid = (long long)blockIdx.x * blockDim.x + threadIdx.x;
  if (gid >= n4) return;
  int c = (int)((gid & 31) << 2);
  float4 x = X[gid];
  float4 y;
  y.x = x.x * scale[c + 0] + shift[c + 0];
  y.y = x.y * scale[c + 1] + shift[c + 1];
  y.z = x.z * scale[c + 2] + shift[c + 2];
  y.w = x.w * scale[c + 3] + shift[c + 3];
  Y[gid] = y;
}

static inline int cdiv(long long a, long long b) { return (int)((a + b - 1) / b); }

extern "C" void kernel_launch(void* const* d_in, const int* in_sizes, int n_in,
                              void* d_out, int out_size, void* d_ws, size_t ws_size,
                              hipStream_t stream) {
  (void)n_in; (void)out_size; (void)ws_size;
  const float* h   = (const float*)d_in[0];
  const float* e   = (const float*)d_in[1];
  const int*   src = (const int*)d_in[2];
  const int*   dst = (const int*)d_in[3];
  const float* Wq = (const float*)d_in[4];
  const float* Wk = (const float*)d_in[5];
  const float* Wv = (const float*)d_in[6];
  const float* We = (const float*)d_in[7];
  const float* Oh_w = (const float*)d_in[8];
  const float* Oh_b = (const float*)d_in[9];
  const float* Oe_w = (const float*)d_in[10];
  const float* Oe_b = (const float*)d_in[11];
  const float* bn1h_g = (const float*)d_in[12]; const float* bn1h_b = (const float*)d_in[13];
  const float* bn1e_g = (const float*)d_in[14]; const float* bn1e_b = (const float*)d_in[15];
  const float* bn2h_g = (const float*)d_in[16]; const float* bn2h_b = (const float*)d_in[17];
  const float* bn2e_g = (const float*)d_in[18]; const float* bn2e_b = (const float*)d_in[19];
  const float* f1h_w = (const float*)d_in[20]; const float* f1h_b = (const float*)d_in[21];
  const float* f2h_w = (const float*)d_in[22]; const float* f2h_b = (const float*)d_in[23];
  const float* f1e_w = (const float*)d_in[24]; const float* f1e_b = (const float*)d_in[25];
  const float* f2e_w = (const float*)d_in[26]; const float* f2e_b = (const float*)d_in[27];

  const int N = in_sizes[0] / 128;
  const int E = in_sizes[1] / 128;
  const long long NC = (long long)N * 128;
  const long long EC = (long long)E * 128;

  // workspace carve
  float* ws = (float*)d_ws;
  float* qh  = ws;
  float* kh  = qh + NC;
  float* vh  = kh + NC;
  float* B   = vh + NC;        // pe -> e_att (in place)
  float* mN  = B + EC;
  float* sN  = mN + NC;
  float* wV  = sN + NC;
  float* h1  = wV + NC;        // h1_pre -> h1 (in place)
  float* h2p = h1 + NC;
  float* e1  = h2p + NC;       // e1_pre -> e1 -> e2_pre (in place)
  float* gsum  = e1 + EC;
  float* gsq   = gsum + 128;
  float* scale = gsq + 128;
  float* shift = scale + 128;
  unsigned short* wbf = (unsigned short*)(shift + 128);
  unsigned short* wq_t  = wbf;
  unsigned short* wk_t  = wq_t + 16384;
  unsigned short* wv_t  = wk_t + 16384;
  unsigned short* we_t  = wv_t + 16384;
  unsigned short* oh_t  = we_t + 16384;
  unsigned short* oe_t  = oh_t + 16384;
  unsigned short* f1h_t = oe_t + 16384;
  unsigned short* f2h_t = f1h_t + 32768;
  unsigned short* f1e_t = f2h_t + 32768;
  unsigned short* f2e_t = f1e_t + 32768;

  // weight conversion
  wt_convert<<<cdiv(16384, 256), 256, 0, stream>>>(Wq, wq_t, 128, 128);
  wt_convert<<<cdiv(16384, 256), 256, 0, stream>>>(Wk, wk_t, 128, 128);
  wt_convert<<<cdiv(16384, 256), 256, 0, stream>>>(Wv, wv_t, 128, 128);
  wt_convert<<<cdiv(16384, 256), 256, 0, stream>>>(We, we_t, 128, 128);
  wt_convert<<<cdiv(16384, 256), 256, 0, stream>>>(Oh_w, oh_t, 128, 128);
  wt_convert<<<cdiv(16384, 256), 256, 0, stream>>>(Oe_w, oe_t, 128, 128);
  wt_convert<<<cdiv(32768, 256), 256, 0, stream>>>(f1h_w, f1h_t, 128, 256);
  wt_convert<<<cdiv(32768, 256), 256, 0, stream>>>(f2h_w, f2h_t, 256, 128);
  wt_convert<<<cdiv(32768, 256), 256, 0, stream>>>(f1e_w, f1e_t, 128, 256);
  wt_convert<<<cdiv(32768, 256), 256, 0, stream>>>(f2e_w, f2e_t, 256, 128);

  // init segment buffers
  fillf4<<<cdiv(NC / 4, 256), 256, 0, stream>>>((float4*)mN, -__builtin_huge_valf(), NC / 4);
  fillf4<<<cdiv(NC / 4, 256), 256, 0, stream>>>((float4*)sN, 0.0f, NC / 4);
  fillf4<<<cdiv(NC / 4, 256), 256, 0, stream>>>((float4*)wV, 0.0f, NC / 4);

  // projections
  int gbN = cdiv(N, 128), gbE = cdiv(E, 128);
  gemm128_bf16<false, false><<<gbN, 256, 0, stream>>>(h, (const unsigned*)wq_t, nullptr, nullptr, qh, N);
  gemm128_bf16<false, false><<<gbN, 256, 0, stream>>>(h, (const unsigned*)wk_t, nullptr, nullptr, kh, N);
  gemm128_bf16<false, false><<<gbN, 256, 0, stream>>>(h, (const unsigned*)wv_t, nullptr, nullptr, vh, N);
  gemm128_bf16<false, false><<<gbE, 256, 0, stream>>>(e, (const unsigned*)we_t, nullptr, nullptr, B, E);

  // attention softmax + aggregation
  int gE = cdiv(EC, 256);
  attn_score<<<gE, 256, 0, stream>>>(kh, qh, B, src, dst, mN, EC);
  attn_expsum<<<gE, 256, 0, stream>>>(B, dst, mN, sN, EC);
  attn_agg<<<gE, 256, 0, stream>>>(B, vh, src, dst, mN, sN, wV, EC);

  // output projections with residual
  gemm128_bf16<true, true><<<gbN, 256, 0, stream>>>(wV, (const unsigned*)oh_t, Oh_b, h, h1, N);
  gemm128_bf16<true, true><<<gbE, 256, 0, stream>>>(B, (const unsigned*)oe_t, Oe_b, e, e1, E);

  // BN1 (in place)
  fillf4<<<1, 64, 0, stream>>>((float4*)gsum, 0.0f, 64);
  bn_stats<<<cdiv(N, 2048), 256, 0, stream>>>(h1, gsum, gsq, N);
  bn_finalize<<<1, 128, 0, stream>>>(gsum, gsq, bn1h_g, bn1h_b, scale, shift, 1.0f / N);
  bn_apply4<<<cdiv(NC / 4, 256), 256, 0, stream>>>((const float4*)h1, scale, shift, (float4*)h1, NC / 4);

  fillf4<<<1, 64, 0, stream>>>((float4*)gsum, 0.0f, 64);
  bn_stats<<<cdiv(E, 2048), 256, 0, stream>>>(e1, gsum, gsq, E);
  bn_finalize<<<1, 128, 0, stream>>>(gsum, gsq, bn1e_g, bn1e_b, scale, shift, 1.0f / E);
  bn_apply4<<<cdiv(EC / 4, 256), 256, 0, stream>>>((const float4*)e1, scale, shift, (float4*)e1, EC / 4);

  // fused FFN with residual
  ffn_fused<<<gbN, 256, 0, stream>>>(h1, (const unsigned*)f1h_t, f1h_b,
                                     (const unsigned*)f2h_t, f2h_b, h2p, N);
  ffn_fused<<<gbE, 256, 0, stream>>>(e1, (const unsigned*)f1e_t, f1e_b,
                                     (const unsigned*)f2e_t, f2e_b, e1, E);

  // BN2 -> d_out
  float* out_h = (float*)d_out;
  float* out_e = out_h + NC;
  fillf4<<<1, 64, 0, stream>>>((float4*)gsum, 0.0f, 64);
  bn_stats<<<cdiv(N, 2048), 256, 0, stream>>>(h2p, gsum, gsq, N);
  bn_finalize<<<1, 128, 0, stream>>>(gsum, gsq, bn2h_g, bn2h_b, scale, shift, 1.0f / N);
  bn_apply4<<<cdiv(NC / 4, 256), 256, 0, stream>>>((const float4*)h2p, scale, shift, (float4*)out_h, NC / 4);

  fillf4<<<1, 64, 0, stream>>>((float4*)gsum, 0.0f, 64);
  bn_stats<<<cdiv(E, 2048), 256, 0, stream>>>(e1, gsum, gsq, E);
  bn_finalize<<<1, 128, 0, stream>>>(gsum, gsq, bn2e_g, bn2e_b, scale, shift, 1.0f / E);
  bn_apply4<<<cdiv(EC / 4, 256), 256, 0, stream>>>((const float4*)e1, scale, shift, (float4*)out_e, EC / 4);
}
